// Attention_67276367725225
// MI455X (gfx1250) — matrix operand order repaired and checked by reading, compiled, not run on hardware
//
#include <hip/hip_runtime.h>

// ---------------------------------------------------------------------------
// MHA layer (RoPE + sliding-window causal attention) for MI455X / gfx1250.
// bf16 WMMA (v_wmma_f32_16x16x32_bf16) everywhere, f32 accumulation.
// Attention K/V slabs staged into LDS by the Tensor Data Mover (TDM),
// double-buffered, synchronized with s_wait_tensorcnt + workgroup barriers.
// ---------------------------------------------------------------------------

typedef __attribute__((ext_vector_type(16))) __bf16 v16bf;
typedef __attribute__((ext_vector_type(8)))  __bf16 v8bf;
typedef __attribute__((ext_vector_type(8)))  float  v8f;
typedef __attribute__((ext_vector_type(4)))  unsigned int v4u;
typedef __attribute__((ext_vector_type(8)))  int v8i;
typedef __attribute__((ext_vector_type(4)))  int v4i;

#define BATCH   4
#define SEQ     2048
#define EMB     1024
#define NHEAD   16
#define HDIM    64
#define WINL    256
#define ROWS    (BATCH*SEQ)      // 8192

__device__ __forceinline__ v8f wmma_bf16(v16bf a, v16bf b, v8f c) {
  // D = A(16x32 bf16) * B(32x16 bf16) + C(16x16 f32)
  return __builtin_amdgcn_wmma_f32_16x16x32_bf16(
      /*neg_a=*/false, a, /*neg_b=*/false, b,
      /*c_mod=*/(short)0, c, /*reuse_a=*/false, /*reuse_b=*/false);
}

// TDM: 2-D tile (tile_d0 contiguous elems x tile_d1 rows, bf16) -> LDS.
// D# group0/group1 bitfields per CDNA5 ISA 8.3/8.4; groups 2/3 zero (2-D).
// This toolchain's builtin takes 6 args (g0,g1,g2,g3,g4-pad,cpol).
__device__ __forceinline__ void tdm_load_2d(unsigned lds_off, const void* gptr,
                                            unsigned tile_d0, unsigned tile_d1,
                                            unsigned long long stride0) {
  unsigned long long ga = (unsigned long long)gptr;
  v4u g0;
  g0[0] = 1u;                                  // count=1, user descriptor
  g0[1] = lds_off;                             // LDS byte address
  g0[2] = (unsigned)ga;                        // global_addr[31:0]
  g0[3] = (unsigned)((ga >> 32) & 0x1FFFFFFu)  // global_addr[56:32]
          | (2u << 30);                        // type=2 ("image")
  v8i g1;
  unsigned td0 = tile_d0, td1 = tile_d1;       // tensor dims = tile (no OOB)
  g1[0] = (int)(1u << 16);                              // data_size=1 (2B)
  g1[1] = (int)((td0 & 0xFFFFu) << 16);                 // tensor_dim0[15:0]
  g1[2] = (int)((td0 >> 16) | ((td1 & 0xFFFFu) << 16)); // dim0 hi | dim1 lo
  g1[3] = (int)((td1 >> 16) | (tile_d0 << 16));         // dim1 hi | tile_dim0
  g1[4] = (int)(tile_d1 & 0xFFFFu);                     // tile_dim1 (dim2=0)
  g1[5] = (int)(stride0 & 0xFFFFFFFFull);               // dim0_stride lo
  g1[6] = (int)((stride0 >> 32) & 0xFFFFull);           // dim0_stride hi
  g1[7] = 0;
  v4i z4 = {0, 0, 0, 0};
  v8i z8 = {0, 0, 0, 0, 0, 0, 0, 0};
  __builtin_amdgcn_tensor_load_to_lds(g0, g1, z4, z4, z8, 0);
}

__device__ __forceinline__ unsigned lds_off_of(const void* p) {
  return (unsigned)(unsigned long long)p;      // flat->LDS: low 32 bits
}

// --------------------------- converters ------------------------------------

__global__ void k_cvt_x(const float* __restrict__ x, __bf16* __restrict__ xb, int n) {
  for (int i = blockIdx.x * blockDim.x + threadIdx.x; i < n;
       i += gridDim.x * blockDim.x)
    xb[i] = (__bf16)x[i];
}

// W [K][N] f32  ->  Wt [N][K] bf16  (so B-fragments are contiguous in K)
__global__ void k_cvt_w(const float* __restrict__ w0, const float* __restrict__ w1,
                        const float* __restrict__ w2, const float* __restrict__ w3,
                        __bf16* __restrict__ o0, __bf16* __restrict__ o1,
                        __bf16* __restrict__ o2, __bf16* __restrict__ o3) {
  const float* w; __bf16* o;
  switch (blockIdx.y) {
    case 0:  w = w0; o = o0; break;
    case 1:  w = w1; o = o1; break;
    case 2:  w = w2; o = o2; break;
    default: w = w3; o = o3; break;
  }
  int idx = blockIdx.x * blockDim.x + threadIdx.x;   // 0 .. EMB*EMB-1
  int n = idx >> 10;          // dst row  (N)
  int k = idx & (EMB - 1);    // dst col  (K)
  o[idx] = (__bf16)w[(size_t)k * EMB + n];
}

// --------------------------- fused QKV GEMM --------------------------------
// grid: (ROWS/128, NHEAD, 3{q,k,v}); block: 256 (8 waves), wave -> 16x64 tile.
__global__ __launch_bounds__(256) void k_gemm_qkv(
    const __bf16* __restrict__ xb,
    const __bf16* __restrict__ wtq, const __bf16* __restrict__ wtk,
    const __bf16* __restrict__ wtv,
    const float* __restrict__ bq,  const float* __restrict__ bv,
    const float* __restrict__ cosp, const float* __restrict__ sinp,
    __bf16* __restrict__ q_r, __bf16* __restrict__ k_r, __bf16* __restrict__ v_t)
{
  const int lane = threadIdx.x & 31;
  const int wave = threadIdx.x >> 5;
  const int l16  = lane & 15;
  const int half = lane >> 4;
  const int m0   = (blockIdx.x * 8 + wave) * 16;
  const int h    = blockIdx.y;
  const int n0   = h * 64;
  const int which = blockIdx.z;          // 0=Q 1=K 2=V
  const __bf16* wt = (which == 0) ? wtq : (which == 1) ? wtk : wtv;

  v8f acc[4] = {v8f{}, v8f{}, v8f{}, v8f{}};
  const __bf16* aptr = xb + (size_t)(m0 + l16) * EMB;   // A row for this lane

  for (int k = 0; k < EMB; k += 32) {
    union { v16bf v; v8bf p[2]; } af;
    // contiguous k order (B same): elems 0..15 -> K = half*16+0..15
    af.v = *(const v16bf*)(aptr + k + half * 16);
    v16bf b0 = *(const v16bf*)(wt + (size_t)(n0 +  0 + l16) * EMB + k + half * 16);
    v16bf b1 = *(const v16bf*)(wt + (size_t)(n0 + 16 + l16) * EMB + k + half * 16);
    v16bf b2 = *(const v16bf*)(wt + (size_t)(n0 + 32 + l16) * EMB + k + half * 16);
    v16bf b3 = *(const v16bf*)(wt + (size_t)(n0 + 48 + l16) * EMB + k + half * 16);
    acc[0] = wmma_bf16(af.v, b0, acc[0]);
    acc[1] = wmma_bf16(af.v, b1, acc[1]);
    acc[2] = wmma_bf16(af.v, b2, acc[2]);
    acc[3] = wmma_bf16(af.v, b3, acc[3]);
  }

  float vals[4][8];
#pragma unroll
  for (int t = 0; t < 4; ++t)
#pragma unroll
    for (int r = 0; r < 8; ++r) vals[t][r] = acc[t][r];

  if (which == 0) {
#pragma unroll
    for (int t = 0; t < 4; ++t) {
      float bb = bq[n0 + 16 * t + l16];
#pragma unroll
      for (int r = 0; r < 8; ++r) vals[t][r] += bb;
    }
  } else if (which == 2) {
#pragma unroll
    for (int t = 0; t < 4; ++t) {
      float bb = bv[n0 + 16 * t + l16];
#pragma unroll
      for (int r = 0; r < 8; ++r) vals[t][r] += bb;
    }
  }

#pragma unroll
  for (int r = 0; r < 8; ++r) {
    int row = m0 + r + 8 * half;       // C layout: VGPR r holds M = r + 8*half
    int b   = row / SEQ;
    int s   = row % SEQ;
    if (which < 2) {
      // RoPE: dims d (tiles 0,1) pair with d+32 (tiles 2,3); cos[d]==cos[d+32]
#pragma unroll
      for (int t = 0; t < 2; ++t) {
        int d = 16 * t + l16;
        float cs = cosp[((size_t)b * SEQ + s) * HDIM + d];
        float sn = sinp[((size_t)b * SEQ + s) * HDIM + d];
        float x1 = vals[t][r], x2 = vals[t + 2][r];
        vals[t][r]     = x1 * cs - x2 * sn;
        vals[t + 2][r] = x2 * cs + x1 * sn;
      }
      __bf16* dst = (which == 0 ? q_r : k_r) +
                    (((size_t)b * NHEAD + h) * SEQ + s) * HDIM;
#pragma unroll
      for (int t = 0; t < 4; ++t) dst[16 * t + l16] = (__bf16)vals[t][r];
    } else {
      // V stored transposed [B,H,D,S] so PV B-fragments are contiguous in S
      __bf16* dst = v_t + (((size_t)b * NHEAD + h) * HDIM) * SEQ;
#pragma unroll
      for (int t = 0; t < 4; ++t)
        dst[(size_t)(16 * t + l16) * SEQ + s] = (__bf16)vals[t][r];
    }
  }
}

// --------------------------- flash attention -------------------------------
// One block = 4 waves = 64 consecutive queries of one (b,h). K/V slabs
// (32 keys) staged into LDS by the TDM, double-buffered; wave 0 drives the
// DMA pipeline (s_wait_tensorcnt), barriers publish each slab.
// q0_wg is a multiple of 64 => slab tile range is even and fully in-bounds.
__global__ __launch_bounds__(128) void k_attn(
    const __bf16* __restrict__ q_r, const __bf16* __restrict__ k_r,
    const __bf16* __restrict__ v_t, const unsigned char* __restrict__ maskp,
    __bf16* __restrict__ ctx)
{
  __shared__ __bf16 kbuf[2][32][64];     // [buf][slab key][dim]   4KB each
  __shared__ __bf16 vbuf[2][64][32];     // [buf][dim][slab key]   4KB each
  __shared__ __bf16 pbuf[4][16][32];     // per-wave P bounce (C->A relayout)

  const int lane = threadIdx.x & 31;
  const int wave = threadIdx.x >> 5;
  const int l16  = lane & 15;
  const int half = lane >> 4;
  const int qwg  = blockIdx.x & 31;            // 64-query group in sequence
  const int h    = (blockIdx.x >> 5) & 15;
  const int b    = blockIdx.x >> 9;
  const int q0   = qwg * 64 + wave * 16;       // this wave's query block
  const float scale = 0.125f;                  // 1/sqrt(64)

  const __bf16* qbase = q_r + (((size_t)b * NHEAD + h) * SEQ + q0) * HDIM;
  const __bf16* kbase = k_r + (((size_t)b * NHEAD + h) * SEQ) * HDIM;
  const __bf16* vbase = v_t + (((size_t)b * NHEAD + h) * HDIM) * SEQ;

  // Q A-fragments for K-chunks c=0,1 (head_dim 64 = 2 x 32)
  union { v16bf v; v8bf p[2]; } qa0, qa1;
  {
    const __bf16* qr = qbase + (size_t)l16 * HDIM;
    qa0.v = *(const v16bf*)(qr + 0 + half * 16);
    qa1.v = *(const v16bf*)(qr + 32 + half * 16);
  }

  v8f o0 = {}, o1 = {}, o2 = {}, o3 = {};
  float m[8], l[8];
#pragma unroll
  for (int r = 0; r < 8; ++r) { m[r] = -1e30f; l[r] = 0.f; }

  // Workgroup-shared slab schedule (union of all 4 windows); always even.
  const int base    = (qwg * 64) >> 4;                 // multiple of 4
  const int ktStart = (base >= 16) ? base - 16 : 0;    // even
  const int ktEnd   = base + 3;
  const int nPairs  = (ktEnd - ktStart + 1) >> 1;

  // Prime the pipeline: slab 0 -> buffer 0 (wave 0 drives the TDM).
  if (wave == 0) {
    int kb0 = ktStart * 16;
    tdm_load_2d(lds_off_of(&kbuf[0][0][0]), kbase + (size_t)kb0 * HDIM,
                /*d0=*/HDIM, /*d1=*/32, /*stride0=*/HDIM);
    tdm_load_2d(lds_off_of(&vbuf[0][0][0]), vbase + kb0,
                /*d0=*/32, /*d1=*/HDIM, /*stride0=*/SEQ);
  }

  for (int p = 0; p < nPairs; ++p) {
    const int kb  = (ktStart + 2 * p) * 16;    // first key of this 32-key slab
    const int buf = p & 1;

    if (wave == 0) __builtin_amdgcn_s_wait_tensorcnt(0);  // slab p landed
    __syncthreads();                                      // publish to block
    if (wave == 0 && p + 1 < nPairs) {                    // prefetch slab p+1
      int kbn = (ktStart + 2 * (p + 1)) * 16;
      int nb  = (p + 1) & 1;
      tdm_load_2d(lds_off_of(&kbuf[nb][0][0]), kbase + (size_t)kbn * HDIM,
                  HDIM, 32, HDIM);
      tdm_load_2d(lds_off_of(&vbuf[nb][0][0]), vbase + kbn,
                  32, HDIM, SEQ);
    }

    // --- QK^T (two 16-key tiles) from LDS ---
    v8f s0 = {}, s1 = {};
    {
      v16bf k00 = *(const v16bf*)(&kbuf[buf][ 0 + l16][ 0 + half * 16]);
      v16bf k01 = *(const v16bf*)(&kbuf[buf][ 0 + l16][32 + half * 16]);
      v16bf k10 = *(const v16bf*)(&kbuf[buf][16 + l16][ 0 + half * 16]);
      v16bf k11 = *(const v16bf*)(&kbuf[buf][16 + l16][32 + half * 16]);
      s0 = wmma_bf16(qa0.v, k00, s0);
      s0 = wmma_bf16(qa1.v, k01, s0);
      s1 = wmma_bf16(qa0.v, k10, s1);
      s1 = wmma_bf16(qa1.v, k11, s1);
    }

    int key0 = kb + l16, key1 = kb + 16 + l16;            // always in [0,S)
    unsigned char mk0 = maskp[(size_t)b * SEQ + key0];
    unsigned char mk1 = maskp[(size_t)b * SEQ + key1];

    float mx[8];
#pragma unroll
    for (int r = 0; r < 8; ++r) {
      int qa = q0 + r + 8 * half;
      bool v0 = mk0 && key0 <= qa && key0 >= qa - WINL;
      bool v1 = mk1 && key1 <= qa && key1 >= qa - WINL;
      float a0 = v0 ? s0[r] * scale : -__builtin_inff();
      float a1 = v1 ? s1[r] * scale : -__builtin_inff();
      s0[r] = a0; s1[r] = a1;
      float t = fmaxf(a0, a1);             // row reduce across 16 lanes of half
      t = fmaxf(t, __shfl_xor(t, 1, 32));
      t = fmaxf(t, __shfl_xor(t, 2, 32));
      t = fmaxf(t, __shfl_xor(t, 4, 32));
      t = fmaxf(t, __shfl_xor(t, 8, 32));
      mx[r] = t;
    }
#pragma unroll
    for (int r = 0; r < 8; ++r) {
      float mn    = fmaxf(m[r], mx[r]);
      float alpha = __expf(m[r] - mn);
      m[r] = mn;
      float p0 = __expf(s0[r] - mn);       // masked (-inf) -> exactly 0
      float p1 = __expf(s1[r] - mn);
      s0[r] = p0; s1[r] = p1;
      float rs = p0 + p1;
      rs += __shfl_xor(rs, 1, 32);
      rs += __shfl_xor(rs, 2, 32);
      rs += __shfl_xor(rs, 4, 32);
      rs += __shfl_xor(rs, 8, 32);
      l[r] = l[r] * alpha + rs;
      o0[r] *= alpha; o1[r] *= alpha; o2[r] *= alpha; o3[r] *= alpha;
    }

    // P: C-layout -> A-layout via wave-private LDS slab
#pragma unroll
    for (int r = 0; r < 8; ++r) {
      pbuf[wave][r + 8 * half][l16]      = (__bf16)s0[r];
      pbuf[wave][r + 8 * half][16 + l16] = (__bf16)s1[r];
    }
    asm volatile("s_wait_dscnt 0x0" ::: "memory");   // CDNA5 split DS counter
    union { v16bf v; v8bf p2[2]; } pa;
    pa.v = *(const v16bf*)(&pbuf[wave][l16][half * 16]);

    // --- PV from LDS (B = V[32 keys x 64 dims], staged as [dim][key]) ---
    {
      v16bf v0 = *(const v16bf*)(&vbuf[buf][ 0 + l16][half * 16]);
      v16bf v1 = *(const v16bf*)(&vbuf[buf][16 + l16][half * 16]);
      v16bf v2 = *(const v16bf*)(&vbuf[buf][32 + l16][half * 16]);
      v16bf v3 = *(const v16bf*)(&vbuf[buf][48 + l16][half * 16]);
      o0 = wmma_bf16(pa.v, v0, o0);
      o1 = wmma_bf16(pa.v, v1, o1);
      o2 = wmma_bf16(pa.v, v2, o2);
      o3 = wmma_bf16(pa.v, v3, o3);
    }
    __syncthreads();   // everyone done with buf before TDM may overwrite it
  }

#pragma unroll
  for (int r = 0; r < 8; ++r) {
    float inv = 1.0f / l[r];
    int s = q0 + r + 8 * half;
    __bf16* dst = ctx + ((size_t)b * SEQ + s) * EMB + h * HDIM;
    dst[ 0 + l16] = (__bf16)(o0[r] * inv);
    dst[16 + l16] = (__bf16)(o1[r] * inv);
    dst[32 + l16] = (__bf16)(o2[r] * inv);
    dst[48 + l16] = (__bf16)(o3[r] * inv);
  }
}

// --------------------------- output projection -----------------------------
__global__ __launch_bounds__(256) void k_out(
    const __bf16* __restrict__ ctx, const __bf16* __restrict__ wto,
    const float* __restrict__ bo, const unsigned char* __restrict__ maskp,
    float* __restrict__ out)
{
  const int lane = threadIdx.x & 31;
  const int wave = threadIdx.x >> 5;
  const int l16  = lane & 15;
  const int half = lane >> 4;
  const int m0   = (blockIdx.x * 8 + wave) * 16;
  const int n0   = blockIdx.y * 64;

  v8f acc[4] = {v8f{}, v8f{}, v8f{}, v8f{}};
  const __bf16* aptr = ctx + (size_t)(m0 + l16) * EMB;
  for (int k = 0; k < EMB; k += 32) {
    union { v16bf v; v8bf p[2]; } af;
    af.v = *(const v16bf*)(aptr + k + half * 16);
    v16bf b0 = *(const v16bf*)(wto + (size_t)(n0 +  0 + l16) * EMB + k + half * 16);
    v16bf b1 = *(const v16bf*)(wto + (size_t)(n0 + 16 + l16) * EMB + k + half * 16);
    v16bf b2 = *(const v16bf*)(wto + (size_t)(n0 + 32 + l16) * EMB + k + half * 16);
    v16bf b3 = *(const v16bf*)(wto + (size_t)(n0 + 48 + l16) * EMB + k + half * 16);
    acc[0] = wmma_bf16(af.v, b0, acc[0]);
    acc[1] = wmma_bf16(af.v, b1, acc[1]);
    acc[2] = wmma_bf16(af.v, b2, acc[2]);
    acc[3] = wmma_bf16(af.v, b3, acc[3]);
  }
  float bias[4];
#pragma unroll
  for (int t = 0; t < 4; ++t) bias[t] = bo[n0 + 16 * t + l16];
#pragma unroll
  for (int r = 0; r < 8; ++r) {
    int row = m0 + r + 8 * half;
    int b = row / SEQ, s = row % SEQ;
    float mv = (float)maskp[(size_t)b * SEQ + s];
#pragma unroll
    for (int t = 0; t < 4; ++t)
      out[(size_t)row * EMB + n0 + 16 * t + l16] = (acc[t][r] + bias[t]) * mv;
  }
}

// --------------------------- launcher --------------------------------------
extern "C" void kernel_launch(void* const* d_in, const int* in_sizes, int n_in,
                              void* d_out, int out_size, void* d_ws, size_t ws_size,
                              hipStream_t stream) {
  const float* x            = (const float*)d_in[0];
  const unsigned char* mask = (const unsigned char*)d_in[1];
  const float* cosp         = (const float*)d_in[2];
  const float* sinp         = (const float*)d_in[3];
  const float* Wq           = (const float*)d_in[4];
  const float* bq           = (const float*)d_in[5];
  const float* Wk           = (const float*)d_in[6];
  const float* Wv           = (const float*)d_in[7];
  const float* bv           = (const float*)d_in[8];
  const float* Wo           = (const float*)d_in[9];
  const float* bo           = (const float*)d_in[10];
  float* out                = (float*)d_out;

  // workspace layout (88 MB total)
  char* ws = (char*)d_ws;
  __bf16* xb  = (__bf16*)(ws);                        // 16 MB x bf16
  __bf16* wtq = (__bf16*)(ws + (16ull << 20));        // 2 MB each, [N][K]
  __bf16* wtk = (__bf16*)(ws + (18ull << 20));
  __bf16* wtv = (__bf16*)(ws + (20ull << 20));
  __bf16* wto = (__bf16*)(ws + (22ull << 20));
  __bf16* q_r = (__bf16*)(ws + (24ull << 20));        // 16 MB [B,H,S,D]
  __bf16* k_r = (__bf16*)(ws + (40ull << 20));        // 16 MB [B,H,S,D]
  __bf16* v_t = (__bf16*)(ws + (56ull << 20));        // 16 MB [B,H,D,S]
  __bf16* ctx = (__bf16*)(ws + (72ull << 20));        // 16 MB [B,S,E]

  k_cvt_x<<<2048, 256, 0, stream>>>(x, xb, ROWS * EMB);
  k_cvt_w<<<dim3(4096, 4), 256, 0, stream>>>(Wq, Wk, Wv, Wo, wtq, wtk, wtv, wto);
  k_gemm_qkv<<<dim3(ROWS / 128, NHEAD, 3), 256, 0, stream>>>(
      xb, wtq, wtk, wtv, bq, bv, cosp, sinp, q_r, k_r, v_t);
  k_attn<<<BATCH * NHEAD * (SEQ / 64), 128, 0, stream>>>(
      q_r, k_r, v_t, mask, ctx);
  k_out<<<dim3(ROWS / 128, NHEAD), 256, 0, stream>>>(ctx, wto, bo, mask, out);
}
